// PatchAttention_83253646066152
// MI455X (gfx1250) — matrix operand order repaired and checked by reading, not compiled, not run on hardware
//
#include <hip/hip_runtime.h>

// ---------------- problem constants ----------------
#define BATCH 4
#define NSEQ  1024
#define DMODEL 768
#define NH    12
#define HD    64
#define TDOUT 2304            // 3*DMODEL
#define MROWS (BATCH * NSEQ)  // 4096
#define KSTEPS (DMODEL / 32)  // 24

// ---------------- WMMA types ----------------
typedef __attribute__((ext_vector_type(16))) __bf16 v16bf;
typedef __attribute__((ext_vector_type(2)))  __bf16 v2bf;
typedef __attribute__((ext_vector_type(8)))  float  v8f;
typedef int v4i __attribute__((vector_size(16)));   // matches builtin param type

struct __attribute__((aligned(16))) Pack16 { unsigned int x, y, z, w; };
struct __attribute__((aligned(16))) Pack32 { Pack16 lo, hi; };

static __device__ __forceinline__ v8f wmma_bf16(const Pack32& a, const Pack32& b, v8f c) {
    return __builtin_amdgcn_wmma_f32_16x16x32_bf16(
        false, __builtin_bit_cast(v16bf, a),
        false, __builtin_bit_cast(v16bf, b),
        (short)0, c, false, false);
}

// fp32 -> bf16 round-to-nearest-even (scalar fallback)
static __device__ __forceinline__ unsigned short f2bf(float f) {
    unsigned int u = __builtin_bit_cast(unsigned int, f);
    u += 0x7FFFu + ((u >> 16) & 1u);
    return (unsigned short)(u >> 16);
}

// packed 2xfp32 -> 2xbf16 in one dword
static __device__ __forceinline__ unsigned int pack2(float lo, float hi) {
#if __has_builtin(__builtin_amdgcn_cvt_pk_bf16_f32)
    v2bf p = __builtin_amdgcn_cvt_pk_bf16_f32(lo, hi);
    return __builtin_bit_cast(unsigned int, p);
#else
    return (unsigned int)f2bf(lo) | ((unsigned int)f2bf(hi) << 16);
#endif
}

// ---------------- async global -> LDS (ASYNCcnt path), gated ----------------
#if __has_builtin(__builtin_amdgcn_global_load_async_to_lds_b128)
#define HAVE_ASYNC_LDS 1
#endif

static __device__ __forceinline__ void cp16_async(const unsigned short* g, unsigned short* l) {
#ifdef HAVE_ASYNC_LDS
    __builtin_amdgcn_global_load_async_to_lds_b128(
        (v4i*)(void*)const_cast<unsigned short*>(g),
        (v4i*)(void*)l, 0, 0);
#else
    *(Pack16*)l = *(const Pack16*)g;
#endif
}

static __device__ __forceinline__ void wait_async_le4() {
#ifdef HAVE_ASYNC_LDS
#if __has_builtin(__builtin_amdgcn_s_wait_asynccnt)
    __builtin_amdgcn_s_wait_asynccnt(4);
#else
    asm volatile("s_wait_asynccnt 0x4" ::: "memory");
#endif
#endif
}
static __device__ __forceinline__ void wait_async_le0() {
#ifdef HAVE_ASYNC_LDS
#if __has_builtin(__builtin_amdgcn_s_wait_asynccnt)
    __builtin_amdgcn_s_wait_asynccnt(0);
#else
    asm volatile("s_wait_asynccnt 0x0" ::: "memory");
#endif
#endif
}

// =====================================================================
// Kernel 0: streaming fp32 -> bf16 conversion (done ONCE, not per tile)
// Each thread: 8 floats in (32B), 8 bf16 out (16B, single b128 store).
// =====================================================================
__global__ void cvt_kernel(const float* __restrict__ src,
                           unsigned int* __restrict__ dst, int n8) {
    const int i = blockIdx.x * blockDim.x + threadIdx.x;
    if (i >= n8) return;
    const float4* s = (const float4*)src + (size_t)i * 2;
    const float4 a = s[0];
    const float4 b = s[1];
    uint4 o;
    o.x = pack2(a.x, a.y);
    o.y = pack2(a.z, a.w);
    o.z = pack2(b.x, b.y);
    o.w = pack2(b.z, b.w);
    ((uint4*)dst)[i] = o;
}

// =====================================================================
// Kernel 1: qkv = xb @ Wb^T (+bias in epilogue), bf16 WMMA GEMM.
// 256 thr (8 waves) -> 128x128 tile, K-step 32, double-buffered LDS
// filled by async global->LDS copies (overlaps next slab with WMMAs).
// Epilogue scatters bf16 into Q[b,h,n,d], K[b,h,n,d], Vt[b,h,d,n].
// =====================================================================
__global__ __launch_bounds__(256) void qkv_kernel(
    const unsigned short* __restrict__ xb, const unsigned short* __restrict__ Wb,
    const float* __restrict__ bias,
    unsigned short* __restrict__ Qb, unsigned short* __restrict__ Kb,
    unsigned short* __restrict__ Vt) {

    __shared__ __align__(32) unsigned short As[2][128][48]; // [buf][m][k]
    __shared__ __align__(32) unsigned short Bs[2][128][48]; // [buf][n][k]

    const int t    = threadIdx.x;
    const int lane = t & 31;
    const int wave = t >> 5;
    const int n16  = lane & 15;
    const int hf   = lane >> 4;
    const int Nbase = blockIdx.x * 128;
    const int Mbase = blockIdx.y * 128;
    const int wm = wave & 3;   // m quadrant (32 rows)
    const int wn = wave >> 2;  // n half (64 cols)

    v8f acc[2][4];
    #pragma unroll
    for (int mt = 0; mt < 2; ++mt)
        #pragma unroll
        for (int nt = 0; nt < 4; ++nt) acc[mt][nt] = (v8f)(0.0f);

    const int srow = t >> 1;          // staged row 0..127
    const int scg  = (t & 1) << 4;    // col group 0 / 16
    const unsigned short* gA = xb + (Mbase + srow) * DMODEL + scg;
    const unsigned short* gB = Wb + (Nbase + srow) * DMODEL + scg;

    // prologue: fill buffer 0
    cp16_async(gA,     &As[0][srow][scg]);
    cp16_async(gA + 8, &As[0][srow][scg + 8]);
    cp16_async(gB,     &Bs[0][srow][scg]);
    cp16_async(gB + 8, &Bs[0][srow][scg + 8]);

    for (int it = 0; it < KSTEPS; ++it) {
        const int buf = it & 1;
        if (it + 1 < KSTEPS) {                 // issue next slab, then
            const int K1 = (it + 1) * 32;      // wait for current (<=4 left)
            cp16_async(gA + K1,     &As[buf ^ 1][srow][scg]);
            cp16_async(gA + K1 + 8, &As[buf ^ 1][srow][scg + 8]);
            cp16_async(gB + K1,     &Bs[buf ^ 1][srow][scg]);
            cp16_async(gB + K1 + 8, &Bs[buf ^ 1][srow][scg + 8]);
            wait_async_le4();
        } else {
            wait_async_le0();
        }
        __syncthreads();

        Pack32 af[2], bfr[4];
        #pragma unroll
        for (int mt = 0; mt < 2; ++mt) {
            const int m = wm * 32 + mt * 16 + n16;
            af[mt].lo = *(const Pack16*)&As[buf][m][hf * 16];
            af[mt].hi = *(const Pack16*)&As[buf][m][hf * 16 + 8];
        }
        #pragma unroll
        for (int nt = 0; nt < 4; ++nt) {
            const int n = wn * 64 + nt * 16 + n16;
            bfr[nt].lo = *(const Pack16*)&Bs[buf][n][hf * 16];
            bfr[nt].hi = *(const Pack16*)&Bs[buf][n][hf * 16 + 8];
        }
        #pragma unroll
        for (int mt = 0; mt < 2; ++mt)
            #pragma unroll
            for (int nt = 0; nt < 4; ++nt)
                acc[mt][nt] = wmma_bf16(af[mt], bfr[nt], acc[mt][nt]);
        __syncthreads();
    }

    // Epilogue: +bias, convert, scatter into Q / K / V^T
    #pragma unroll
    for (int mt = 0; mt < 2; ++mt) {
        #pragma unroll
        for (int nt = 0; nt < 4; ++nt) {
            const int o  = Nbase + wn * 64 + nt * 16 + n16; // output channel
            const float bv = bias[o];
            #pragma unroll
            for (int r = 0; r < 8; ++r) {
                const int gr   = Mbase + wm * 32 + mt * 16 + hf * 8 + r;
                const int bidx = gr >> 10;
                const int nseq = gr & 1023;
                const unsigned short bfv = f2bf(acc[mt][nt][r] + bv);
                if (o < DMODEL) {
                    const int hh = o >> 6, d = o & 63;
                    Qb[((bidx * NH + hh) * NSEQ + nseq) * HD + d] = bfv;
                } else if (o < 2 * DMODEL) {
                    const int o2 = o - DMODEL, hh = o2 >> 6, d = o2 & 63;
                    Kb[((bidx * NH + hh) * NSEQ + nseq) * HD + d] = bfv;
                } else {
                    const int o2 = o - 2 * DMODEL, hh = o2 >> 6, d = o2 & 63;
                    Vt[((bidx * NH + hh) * HD + d) * NSEQ + nseq] = bfv;
                }
            }
        }
    }
}

// =====================================================================
// Kernel 2: proj[b,h,n] = sum_c coords[b,n,c] * rel_weight[h,c]
// =====================================================================
__global__ void proj_kernel(const float* __restrict__ coords,
                            const float* __restrict__ rel,
                            float* __restrict__ proj) {
    const int idx = blockIdx.x * blockDim.x + threadIdx.x;
    if (idx >= BATCH * NH * NSEQ) return;
    const int n  = idx % NSEQ;
    const int hh = (idx / NSEQ) % NH;
    const int b  = idx / (NSEQ * NH);
    const float* c = coords + (b * NSEQ + n) * 3;
    const float* w = rel + hh * 3;
    proj[idx] = c[0] * w[0] + c[1] * w[1] + c[2] * w[2];
}

// =====================================================================
// Kernel 3: flash attention with separable coord bias.
// Block = 128 thr (4 waves); wave owns 16 q-rows of one (b,h).
// Key loop step 32: 4 score WMMAs + 4 P*V WMMAs per step.
// K/V are L2-resident (12.6 MB bf16 << 192 MB): stream B-operands
// straight from global as contiguous 32B/lane loads.
// =====================================================================
__global__ __launch_bounds__(128) void attn_kernel(
    const unsigned short* __restrict__ Qb, const unsigned short* __restrict__ Kb,
    const unsigned short* __restrict__ Vt, const float* __restrict__ proj,
    float* __restrict__ out) {

    const int lane = threadIdx.x & 31;
    const int wave = threadIdx.x >> 5;
    const int n16  = lane & 15;
    const int hf   = lane >> 4;
    const int bh   = blockIdx.y;
    const int b    = bh / NH;
    const int h    = bh % NH;
    const int qbase = blockIdx.x * 64 + wave * 16;
    const float SCALE = 0.125f; // 1/sqrt(64)

    __shared__ __align__(32) unsigned short Pl[4][16][32]; // per-wave P tile

    // Q A-fragments (16 rows x 64 d), two K=32 chunks
    Pack32 aQ[2];
    {
        const unsigned short* qp = Qb + (bh * NSEQ + qbase + n16) * HD;
        #pragma unroll
        for (int kc = 0; kc < 2; ++kc) {
            aQ[kc].lo = *(const Pack16*)(qp + kc * 32 + hf * 16);
            aQ[kc].hi = *(const Pack16*)(qp + kc * 32 + hf * 16 + 8);
        }
    }
    float pq[8];
    #pragma unroll
    for (int r = 0; r < 8; ++r) pq[r] = proj[bh * NSEQ + qbase + hf * 8 + r];

    float mrow[8], lrow[8];
    #pragma unroll
    for (int r = 0; r < 8; ++r) { mrow[r] = -1e30f; lrow[r] = 0.0f; }
    v8f acc[4];
    #pragma unroll
    for (int ct = 0; ct < 4; ++ct) acc[ct] = (v8f)(0.0f);

    for (int kb = 0; kb < NSEQ; kb += 32) {
        const float pk0 = proj[bh * NSEQ + kb + n16];
        const float pk1 = proj[bh * NSEQ + kb + 16 + n16];

        // scores S = Q * K^T  (16 x 32, two 16x16 col tiles)
        v8f S0 = (v8f)(0.0f), S1 = (v8f)(0.0f);
        #pragma unroll
        for (int kc = 0; kc < 2; ++kc) {
            Pack32 bk0, bk1;
            const unsigned short* kp0 = Kb + (bh * NSEQ + kb + n16) * HD + kc * 32 + hf * 16;
            const unsigned short* kp1 = Kb + (bh * NSEQ + kb + 16 + n16) * HD + kc * 32 + hf * 16;
            bk0.lo = *(const Pack16*)kp0;       bk0.hi = *(const Pack16*)(kp0 + 8);
            bk1.lo = *(const Pack16*)kp1;       bk1.hi = *(const Pack16*)(kp1 + 8);
            S0 = wmma_bf16(aQ[kc], bk0, S0);
            S1 = wmma_bf16(aQ[kc], bk1, S1);
        }

        // online softmax with bias pq[i] - pk[j]
        #pragma unroll
        for (int r = 0; r < 8; ++r) {
            float t0 = S0[r] * SCALE + pq[r] - pk0;
            float t1 = S1[r] * SCALE + pq[r] - pk1;
            float rm = fmaxf(t0, t1);
            #pragma unroll
            for (int mk = 1; mk < 16; mk <<= 1) rm = fmaxf(rm, __shfl_xor(rm, mk, 32));
            const float mnew = fmaxf(mrow[r], rm);
            const float al = __expf(mrow[r] - mnew);
            const float p0 = __expf(t0 - mnew);
            const float p1 = __expf(t1 - mnew);
            float rs = p0 + p1;
            #pragma unroll
            for (int mk = 1; mk < 16; mk <<= 1) rs += __shfl_xor(rs, mk, 32);
            lrow[r] = lrow[r] * al + rs;
            mrow[r] = mnew;
            #pragma unroll
            for (int ct = 0; ct < 4; ++ct) acc[ct][r] *= al;
            Pl[wave][hf * 8 + r][n16]      = f2bf(p0);
            Pl[wave][hf * 8 + r][16 + n16] = f2bf(p1);
        }

        // C-layout -> A-layout via per-wave LDS tile (same-wave LDS is in-order)
        Pack32 aP;
        aP.lo = *(const Pack16*)&Pl[wave][n16][hf * 16];
        aP.hi = *(const Pack16*)&Pl[wave][n16][hf * 16 + 8];

        // O += P * V  (B operand from V^T: contiguous over keys)
        #pragma unroll
        for (int ct = 0; ct < 4; ++ct) {
            const unsigned short* vp =
                Vt + (bh * HD + ct * 16 + n16) * NSEQ + kb + hf * 16;
            Pack32 bv;
            bv.lo = *(const Pack16*)vp;
            bv.hi = *(const Pack16*)(vp + 8);
            acc[ct] = wmma_bf16(aP, bv, acc[ct]);
        }
    }

    // normalize and write (B, N, D) with head interleave
    #pragma unroll
    for (int ct = 0; ct < 4; ++ct) {
        #pragma unroll
        for (int r = 0; r < 8; ++r) {
            const int row = qbase + hf * 8 + r;
            const int col = h * HD + ct * 16 + n16;
            out[(b * NSEQ + row) * DMODEL + col] = acc[ct][r] / lrow[r];
        }
    }
}

// =====================================================================
extern "C" void kernel_launch(void* const* d_in, const int* in_sizes, int n_in,
                              void* d_out, int out_size, void* d_ws, size_t ws_size,
                              hipStream_t stream) {
    (void)in_sizes; (void)n_in; (void)out_size; (void)ws_size;
    const float* x      = (const float*)d_in[0];
    const float* coords = (const float*)d_in[1];
    const float* Wqkv   = (const float*)d_in[2];
    const float* bqkv   = (const float*)d_in[3];
    const float* relw   = (const float*)d_in[4];
    float* out = (float*)d_out;

    const size_t nX = (size_t)MROWS * DMODEL;          // 3,145,728
    const size_t nW = (size_t)TDOUT * DMODEL;          // 1,769,472
    const size_t nQ = (size_t)BATCH * NH * NSEQ * HD;  // 3,145,728

    unsigned short* xb = (unsigned short*)d_ws;
    unsigned short* Wb = xb + nX;
    unsigned short* Qb = Wb + nW;
    unsigned short* Kb = Qb + nQ;
    unsigned short* Vt = Kb + nQ;
    float* proj = (float*)(Vt + nQ);                   // total ~29 MB

    cvt_kernel<<<(int)(nX / 8 / 256), 256, 0, stream>>>(x, (unsigned int*)xb, (int)(nX / 8));
    cvt_kernel<<<(int)(nW / 8 / 256), 256, 0, stream>>>(Wqkv, (unsigned int*)Wb, (int)(nW / 8));
    qkv_kernel<<<dim3(TDOUT / 128, MROWS / 128), 256, 0, stream>>>(
        xb, Wb, bqkv, Qb, Kb, Vt);
    proj_kernel<<<(BATCH * NH * NSEQ + 255) / 256, 256, 0, stream>>>(
        coords, relw, proj);
    attn_kernel<<<dim3(NSEQ / 64, BATCH * NH), 128, 0, stream>>>(
        Qb, Kb, Vt, proj, out);
}
